// BinPackingGAT_80075370266923
// MI455X (gfx1250) — matrix operand order repaired and checked by reading, not compiled, not run on hardware
//
#include <hip/hip_runtime.h>
#include <hip/hip_bf16.h>
#include <math.h>

typedef __attribute__((ext_vector_type(16))) _Float16 v16h;
typedef __attribute__((ext_vector_type(8)))  _Float16 v8h;
typedef __attribute__((ext_vector_type(8)))  float    v8f;
typedef __attribute__((ext_vector_type(4)))  float    v4f;

#define NEG_SLOPE 0.2f

__device__ __forceinline__ void atomicMaxF32(float* addr, float val) {
  // ordered-int trick: correct for any sign mix when initialized to -inf
  if (val >= 0.0f) atomicMax((int*)addr, __float_as_int(val));
  else             atomicMin((unsigned int*)addr, __float_as_uint(val));
}

// ---------------------------------------------------------------------------
// Register-blocked WMMA GEMM: C[M,Nout] = A[M,K](f16,row-major) * W[K,Nout].
// W pre-transposed+converted: Bt[n*K + k] (f16) so a B fragment per lane is
// two 8-value k-runs at 8h and 16+8h (ISA 16-bit B 32x16 layout). One wave computes a
// 16 x (NT*16) strip: per k-step the A fragment is loaded once and fed to NT
// independent-accumulator v_wmma_f32_16x16x32_f16 ops (no D->A/B hazards).
// ---------------------------------------------------------------------------
template<int K, int NT>
__global__ __launch_bounds__(256) void gemm_wmma(
    const _Float16* __restrict__ A, int lda,
    const _Float16* __restrict__ Bt,
    const float* __restrict__ bias,                   // nullable
    float* __restrict__ C32,                          // nullable, ldc = Nout
    _Float16* __restrict__ C16, int ldc16, int coff,  // nullable
    int M, int Nout, int relu)
{
  const int lane = threadIdx.x & 31;
  const int wid  = threadIdx.x >> 5;
  const int mtiles = M >> 4;
  const int mt = blockIdx.x * 8 + wid;
  if (mt >= mtiles) return;                // wave-uniform: EXEC all-ones in WMMA
  const int half = lane >> 4;              // 0: lanes 0-15, 1: lanes 16-31
  const int r = lane & 15;

  const _Float16* arow = A  + (size_t)(mt * 16 + r) * lda;
  const _Float16* bcol = Bt + (size_t)r * K;          // + j*16*K per N-tile
  const int alo   = half ? 8  : 0;         // A: K lo-range base per half-wave
  const int bbase = half ? 8  : 0;         // B: K lo-range base per half-wave

  v8f acc[NT];
#pragma unroll
  for (int j = 0; j < NT; ++j) acc[j] = (v8f){};

#pragma unroll
  for (int kt = 0; kt < K; kt += 32) {
    v8h a0 = *(const v8h*)(arow + kt + alo);        // k = kt+alo .. +7
    v8h a1 = *(const v8h*)(arow + kt + alo + 16);   // k = kt+alo+16 .. +23
    v16h af;
#pragma unroll
    for (int i = 0; i < 8; ++i) { af[i] = a0[i]; af[8 + i] = a1[i]; }
#pragma unroll
    for (int j = 0; j < NT; ++j) {
      const _Float16* bp = bcol + (size_t)j * 16 * K + kt + bbase;
      v8h b0 = *(const v8h*)(bp);
      v8h b1 = *(const v8h*)(bp + 16);
      v16h bf;
#pragma unroll
      for (int i = 0; i < 8; ++i) { bf[i] = b0[i]; bf[8 + i] = b1[i]; }
      acc[j] = __builtin_amdgcn_wmma_f32_16x16x32_f16(
          false, af, false, bf, (short)0, acc[j], false, false);
    }
  }

  // C layout: VGPR i -> row = mt*16 + half*8 + i, col = j*16 + lane%16
#pragma unroll
  for (int j = 0; j < NT; ++j) {
    const int col = j * 16 + r;
    const float bv = bias ? bias[col] : 0.0f;
#pragma unroll
    for (int i = 0; i < 8; ++i) {
      const int row = mt * 16 + half * 8 + i;
      float v = acc[j][i] + bv;
      if (relu) v = v > 0.0f ? v : 0.0f;
      if (C32) C32[(size_t)row * Nout + col] = v;
      if (C16) C16[(size_t)row * ldc16 + coff + col] = (_Float16)v;
    }
  }
}

// W f32 [K x Nout] row-major  ->  Wt f16 [Nout x K]
__global__ void pack_wt(const float* __restrict__ W, _Float16* __restrict__ Wt,
                        int K, int Nout) {
  int t = blockIdx.x * blockDim.x + threadIdx.x;
  if (t >= K * Nout) return;
  int k = t / Nout, n = t % Nout;
  Wt[(size_t)n * K + k] = (_Float16)W[t];
}

__global__ void f32_to_f16(const float* __restrict__ in, _Float16* __restrict__ out, int n) {
  int t = blockIdx.x * blockDim.x + threadIdx.x;
  if (t < n) out[t] = (_Float16)in[t];
}

// es[i] = h[i,:].a_src ; ed[i] = h[i,:].a_dst  (one wave/node, float4 lanes)
__global__ void scores_kernel(const float* __restrict__ h,
                              const float* __restrict__ asrc,
                              const float* __restrict__ adst,
                              float* __restrict__ es, float* __restrict__ ed, int N) {
  int w = (blockIdx.x * blockDim.x + threadIdx.x) >> 5;
  int lane = threadIdx.x & 31;
  if (w >= N) return;
  const v4f* hr = (const v4f*)(h + (size_t)w * 128);
  const v4f* a4 = (const v4f*)asrc;
  const v4f* d4 = (const v4f*)adst;
  v4f hv = hr[lane], av = a4[lane], dv = d4[lane];
  float s = hv.x * av.x + hv.y * av.y + hv.z * av.z + hv.w * av.w;
  float d = hv.x * dv.x + hv.y * dv.y + hv.z * dv.z + hv.w * dv.w;
  for (int o = 16; o > 0; o >>= 1) {
    s += __shfl_xor(s, o, 32);
    d += __shfl_xor(d, o, 32);
  }
  if (lane == 0) { es[w] = s; ed[w] = d; }
}

__global__ void init_seg(float* __restrict__ m, float* __restrict__ denom,
                         float* __restrict__ agg, int N, int C) {
  int t = blockIdx.x * blockDim.x + threadIdx.x;
  if (t < N * C) agg[t] = 0.0f;
  if (t < N) { m[t] = -__builtin_inff(); denom[t] = 0.0f; }
}

__device__ __forceinline__ void edge_decode(const int* __restrict__ ei, int E, int e,
                                            int& s, int& d) {
  if (e < E) { s = ei[e]; d = ei[E + e]; }
  else       { s = d = e - E; }            // self loops appended at the end
}

__global__ void edge_logits_max(const int* __restrict__ ei, int E, int N,
                                const float* __restrict__ es, const float* __restrict__ ed,
                                float* __restrict__ logits, float* __restrict__ m) {
  int e = blockIdx.x * blockDim.x + threadIdx.x;
  if (e >= E + N) return;
  int s, d; edge_decode(ei, E, e, s, d);
  float l = es[s] + ed[d];
  l = l > 0.0f ? l : NEG_SLOPE * l;        // leaky relu
  logits[e] = l;
  atomicMaxF32(&m[d], l);
}

__global__ void edge_exp_sum(const int* __restrict__ ei, int E, int N,
                             const float* __restrict__ logits, const float* __restrict__ m,
                             float* __restrict__ p, float* __restrict__ denom) {
  int e = blockIdx.x * blockDim.x + threadIdx.x;
  if (e >= E + N) return;
  int s, d; edge_decode(ei, E, e, s, d);
  float v = __expf(logits[e] - m[d]);
  p[e] = v;
  atomicAdd(&denom[d], v);
}

// agg[dst,:] += (p[e]/denom[dst]) * h[src,:]  (one wave/edge; float4 gather,
// 4 f32 atomics per lane; h and agg are L2-resident at 25.6 MB each)
__global__ void aggregate(const int* __restrict__ ei, int E, int N,
                          const float* __restrict__ p, const float* __restrict__ denom,
                          const float* __restrict__ h, float* __restrict__ agg) {
  int e = (blockIdx.x * blockDim.x + threadIdx.x) >> 5;
  int lane = threadIdx.x & 31;
  if (e >= E + N) return;
  int s, d; edge_decode(ei, E, e, s, d);
  float alpha = p[e] / denom[d];
  v4f hv = ((const v4f*)(h + (size_t)s * 128))[lane];
  float* od = agg + (size_t)d * 128 + lane * 4;
  atomicAdd(od + 0, alpha * hv.x);
  atomicAdd(od + 1, alpha * hv.y);
  atomicAdd(od + 2, alpha * hv.z);
  atomicAdd(od + 3, alpha * hv.w);
}

// layer epilogue: (agg + b) -> relu -> f16, written with stride/offset (concat)
__global__ void bias_relu16(const float* __restrict__ agg, const float* __restrict__ b,
                            _Float16* __restrict__ out, int N, int C, int ldo, int coff) {
  int t = blockIdx.x * blockDim.x + threadIdx.x;
  if (t >= N * C) return;
  int r = t / C, c = t % C;
  float v = agg[t] + b[c];
  v = v > 0.0f ? v : 0.0f;
  out[(size_t)r * ldo + coff + c] = (_Float16)v;
}

__global__ void gather_out(const float* __restrict__ full, const int* __restrict__ idx,
                           float* __restrict__ out, int nIdx, int C) {
  int t = blockIdx.x * blockDim.x + threadIdx.x;   // over nIdx * C/4 float4s
  int c4 = C >> 2;
  if (t >= nIdx * c4) return;
  int i = t / c4, c = t % c4;
  ((v4f*)out)[t] = ((const v4f*)full)[(size_t)idx[i] * c4 + c];
}

extern "C" void kernel_launch(void* const* d_in, const int* in_sizes, int n_in,
                              void* d_out, int out_size, void* d_ws, size_t ws_size,
                              hipStream_t stream) {
  (void)n_in; (void)out_size; (void)ws_size;
  const int N = 50000, C = 128, H = 128, COUT = 64;
  const float* x   = (const float*)d_in[0];
  const int*   eix = (const int*)d_in[1];
  const int*   eiy = (const int*)d_in[2];
  const int*   idx = (const int*)d_in[3];
  const int E    = in_sizes[1] / 2;
  const int nIdx = in_sizes[3];
  const int Et   = E + N;

  const float* Wl[4]  = {(const float*)d_in[4],  (const float*)d_in[8],
                         (const float*)d_in[12], (const float*)d_in[16]};
  const float* asl[4] = {(const float*)d_in[5],  (const float*)d_in[9],
                         (const float*)d_in[13], (const float*)d_in[17]};
  const float* adl[4] = {(const float*)d_in[6],  (const float*)d_in[10],
                         (const float*)d_in[14], (const float*)d_in[18]};
  const float* bl[4]  = {(const float*)d_in[7],  (const float*)d_in[11],
                         (const float*)d_in[15], (const float*)d_in[19]};
  const float* Wfc  = (const float*)d_in[20];
  const float* bfc  = (const float*)d_in[21];
  const float* Wout = (const float*)d_in[22];
  const float* bout = (const float*)d_in[23];

  // ---- carve workspace (256B aligned) ----
  size_t off = 0;
  auto carve = [&](size_t bytes) -> void* {
    void* p = (char*)d_ws + off;
    off += (bytes + 255) & ~(size_t)255;
    return p;
  };
  _Float16* Wt[4];
  for (int i = 0; i < 4; ++i) Wt[i] = (_Float16*)carve(128 * 128 * sizeof(_Float16));
  _Float16* Wtfc  = (_Float16*)carve(128 * 256 * sizeof(_Float16)); // [n=128][k=256]
  _Float16* Wtout = (_Float16*)carve(64 * 128 * sizeof(_Float16));  // [n=64][k=128]
  _Float16* X16  = (_Float16*)carve((size_t)N * C * sizeof(_Float16));
  _Float16* T16  = (_Float16*)carve((size_t)N * H * sizeof(_Float16));
  _Float16* CC16 = (_Float16*)carve((size_t)N * 2 * H * sizeof(_Float16)); // concat
  float* h32  = (float*)carve((size_t)N * H * sizeof(float));
  float* agg  = (float*)carve((size_t)N * H * sizeof(float));
  float* es   = (float*)carve((size_t)N * sizeof(float));
  float* ed   = (float*)carve((size_t)N * sizeof(float));
  float* mseg = (float*)carve((size_t)N * sizeof(float));
  float* dseg = (float*)carve((size_t)N * sizeof(float));
  float* lg   = (float*)carve((size_t)Et * sizeof(float));
  float* pe   = (float*)carve((size_t)Et * sizeof(float));
  float* full = (float*)carve((size_t)N * COUT * sizeof(float));

  // ---- one-time packs (cheap) ----
  for (int i = 0; i < 4; ++i)
    pack_wt<<<(128 * 128 + 255) / 256, 256, 0, stream>>>(Wl[i], Wt[i], 128, 128);
  pack_wt<<<(256 * 128 + 255) / 256, 256, 0, stream>>>(Wfc,  Wtfc,  256, 128);
  pack_wt<<<(128 * 64  + 255) / 256, 256, 0, stream>>>(Wout, Wtout, 128, 64);
  f32_to_f16<<<(N * C + 255) / 256, 256, 0, stream>>>(x, X16, N * C);

  const int mwaves = N / 16;               // 3125 wave-strips of 16 rows

  auto gat_layer = [&](const _Float16* Ain, const _Float16* Wtl,
                       const float* as_, const float* ad_, const float* b_,
                       const int* ei, _Float16* out16, int ldo, int coff) {
    gemm_wmma<128, 8><<<(mwaves + 7) / 8, 256, 0, stream>>>(
        Ain, 128, Wtl, nullptr, h32, nullptr, 0, 0, N, H, 0);
    scores_kernel<<<(N + 7) / 8, 256, 0, stream>>>(h32, as_, ad_, es, ed, N);
    init_seg<<<(N * H + 255) / 256, 256, 0, stream>>>(mseg, dseg, agg, N, H);
    edge_logits_max<<<(Et + 255) / 256, 256, 0, stream>>>(ei, E, N, es, ed, lg, mseg);
    edge_exp_sum<<<(Et + 255) / 256, 256, 0, stream>>>(ei, E, N, lg, mseg, pe, dseg);
    aggregate<<<(Et + 7) / 8, 256, 0, stream>>>(ei, E, N, pe, dseg, h32, agg);
    bias_relu16<<<(N * H + 255) / 256, 256, 0, stream>>>(agg, b_, out16, N, H, ldo, coff);
  };

  // branch x: layer1 -> T16, layer2 -> concat[:, 0:128]
  gat_layer(X16, Wt[0], asl[0], adl[0], bl[0], eix, T16, 128, 0);
  gat_layer(T16, Wt[1], asl[1], adl[1], bl[1], eix, CC16, 256, 0);
  // branch y: layer1 -> T16 (reuse), layer2 -> concat[:, 128:256]
  gat_layer(X16, Wt[2], asl[2], adl[2], bl[2], eiy, T16, 128, 0);
  gat_layer(T16, Wt[3], asl[3], adl[3], bl[3], eiy, CC16, 256, 128);

  // FC: [N,256] @ [256,128] + bfc, relu -> T16 (f16)
  gemm_wmma<256, 8><<<(mwaves + 7) / 8, 256, 0, stream>>>(
      CC16, 256, Wtfc, bfc, nullptr, T16, 128, 0, N, 128, 1);
  // OUT: [N,128] @ [128,64] + bout -> full (f32)
  gemm_wmma<128, 4><<<(mwaves + 7) / 8, 256, 0, stream>>>(
      T16, 128, Wtout, bout, full, nullptr, 0, 0, N, 64, 0);

  gather_out<<<(nIdx * (COUT / 4) + 255) / 256, 256, 0, stream>>>(
      full, idx, (float*)d_out, nIdx, COUT);
}
